// IntegratedDCALLM_40269613367685
// MI455X (gfx1250) — matrix operand order repaired and checked by reading, not compiled, not run on hardware
//
#include <hip/hip_runtime.h>
#include <hip/hip_bf16.h>
#include <math.h>

typedef __attribute__((ext_vector_type(16))) _Float16 v16h;
typedef __attribute__((ext_vector_type(8)))  float    v8f;

namespace cfg {
constexpr int B = 2, S = 2048, D = 256, H = 4, L = 4, V = 50257;
constexpr int HD = D / H;        // 64
constexpr int MROWS = B * S;     // 4096
}

// ---------------------------------------------------------------------------
// WMMA helpers (CDNA5 wave32, 16x16x32 f16 -> f32)
// ---------------------------------------------------------------------------
__device__ __forceinline__ v8f wmma_f16(v16h a, v16h b, v8f c) {
  return __builtin_amdgcn_wmma_f32_16x16x32_f16(false, a, false, b, (short)0, c,
                                                false, false);
}

// A fragment (16x32 f16). Contiguous-k layout used here: lane -> m = lane&15, half = lane>>4;
// vgpr v holds k = half*16 + 2v (+0/1). LDS tile is m-major.
__device__ __forceinline__ v16h load_a_frag(const _Float16* lds, int row0,
                                            int k0, int ld) {
  const int lane = threadIdx.x & 31;
  const int m = row0 + (lane & 15);
  const int h = lane >> 4;
  v16h f;
#pragma unroll
  for (int v = 0; v < 8; ++v) {
    const int k = k0 + (h << 4) + (v << 1);
    f[2 * v]     = lds[m * ld + k];
    f[2 * v + 1] = lds[m * ld + k + 1];
  }
  return f;
}

// B fragment (32x16 f16) from a k-major LDS tile: value(k,n) = lds[k*ld+n].
// Contiguous-k layout used here: lane -> n = lane&15, half = lane>>4; vgpr v: k = half*16 + 2v.
__device__ __forceinline__ v16h load_b_frag(const _Float16* lds, int k0,
                                            int n0, int ld) {
  const int lane = threadIdx.x & 31;
  const int n = n0 + (lane & 15);
  const int h = lane >> 4;
  v16h f;
#pragma unroll
  for (int v = 0; v < 8; ++v) {
    const int k = k0 + (h << 4) + (v << 1);
    f[2 * v]     = lds[k * ld + n];
    f[2 * v + 1] = lds[(k + 1) * ld + n];
  }
  return f;
}

// B fragment from an n-major LDS tile: value(k,n) = lds[n*ld+k]  (for Q@K^T).
__device__ __forceinline__ v16h load_bT_frag(const _Float16* lds, int k0,
                                             int n0, int ld) {
  const int lane = threadIdx.x & 31;
  const int n = n0 + (lane & 15);
  const int h = lane >> 4;
  v16h f;
#pragma unroll
  for (int v = 0; v < 8; ++v) {
    const int k = k0 + (h << 4) + (v << 1);
    f[2 * v]     = lds[n * ld + k];
    f[2 * v + 1] = lds[n * ld + k + 1];
  }
  return f;
}

// ---------------------------------------------------------------------------
// Embedding: h[b,s,:] = tok_emb[ids[b,s],:] + pos_emb[s,:]
// ---------------------------------------------------------------------------
__global__ __launch_bounds__(256) void embed_kernel(
    const int* __restrict__ ids, const float* __restrict__ tok,
    const float* __restrict__ pos, float* __restrict__ h) {
  using namespace cfg;
  const int row = blockIdx.x;           // [0, B*S)
  const int t = threadIdx.x;            // [0, D)
  const int s = row & (S - 1);
  const int id = ids[row];
  h[(size_t)row * D + t] = tok[(size_t)id * D + t] + pos[(size_t)s * D + t];
}

// ---------------------------------------------------------------------------
// LayerNorm over D=256 (one row per block, 256 threads)
// ---------------------------------------------------------------------------
__global__ __launch_bounds__(256) void layernorm_kernel(
    const float* __restrict__ x, const float* __restrict__ sc,
    const float* __restrict__ bi, float* __restrict__ y) {
  using namespace cfg;
  const int row = blockIdx.x, t = threadIdx.x;
  __shared__ float red[256];
  const float v = x[(size_t)row * D + t];
  red[t] = v;
  __syncthreads();
  for (int st = 128; st > 0; st >>= 1) {
    if (t < st) red[t] += red[t + st];
    __syncthreads();
  }
  const float mean = red[0] * (1.f / D);
  __syncthreads();
  const float d = v - mean;
  red[t] = d * d;
  __syncthreads();
  for (int st = 128; st > 0; st >>= 1) {
    if (t < st) red[t] += red[t + st];
    __syncthreads();
  }
  const float var = red[0] * (1.f / D);
  y[(size_t)row * D + t] = d * rsqrtf(var + 1e-5f) * sc[t] + bi[t];
}

// ---------------------------------------------------------------------------
// WMMA GEMM: C[M,N] = A[M,K] @ B + bias (+ epilogue)
//   TRANSB=false: B is K x N row-major;  TRANSB=true: B is N x K row-major
//   EPI: 0 = bias only, 1 = bias + exact GELU, 2 = bias + residual add
// 64x64 block tile, 128 threads (4 waves), BK=32, fp32->fp16 staged in LDS.
// M is always a multiple of 64; K a multiple of 32; N bounds-checked.
// ---------------------------------------------------------------------------
template <bool TRANSB, int EPI>
__global__ __launch_bounds__(128) void gemm_kernel(
    const float* __restrict__ A, const float* __restrict__ Bm,
    const float* __restrict__ bias, const float* __restrict__ res,
    float* __restrict__ C, int M, int N, int K) {
  __shared__ _Float16 As[64 * 32];
  __shared__ _Float16 Bs[32 * 64];
  const int tid = threadIdx.x;
  const int wave = tid >> 5, lane = tid & 31;
  const int ln16 = lane & 15, lhalf = lane >> 4;
  const int wr = wave >> 1, wc = wave & 1;
  const int m0 = blockIdx.y * 64, n0 = blockIdx.x * 64;

  v8f acc[2][2];
#pragma unroll
  for (int i = 0; i < 2; ++i)
#pragma unroll
    for (int j = 0; j < 2; ++j)
#pragma unroll
      for (int e = 0; e < 8; ++e) acc[i][j][e] = 0.f;

  for (int kt = 0; kt < K; kt += 32) {
    __syncthreads();
    // stage A (64x32), fp32 -> fp16
    for (int idx = tid; idx < 64 * 32; idx += 128) {
      const int r = idx >> 5, c = idx & 31;
      As[idx] = (_Float16)A[(size_t)(m0 + r) * K + kt + c];
    }
    // stage B (32x64), fp32 -> fp16, N bounds-checked
    for (int idx = tid; idx < 32 * 64; idx += 128) {
      const int r = idx >> 6, c = idx & 63;
      const int gn = n0 + c;
      float x = 0.f;
      if (gn < N)
        x = TRANSB ? Bm[(size_t)gn * K + kt + r] : Bm[(size_t)(kt + r) * N + gn];
      Bs[idx] = (_Float16)x;
    }
    if (kt + 32 < K)  // hint the next A tile toward the caches
      __builtin_prefetch(&A[(size_t)(m0 + (tid >> 1)) * K + kt + 32], 0, 1);
    __syncthreads();

    const v16h af0 = load_a_frag(As, wr * 32, 0, 32);
    const v16h af1 = load_a_frag(As, wr * 32 + 16, 0, 32);
    const v16h bf0 = load_b_frag(Bs, 0, wc * 32, 64);
    const v16h bf1 = load_b_frag(Bs, 0, wc * 32 + 16, 64);
    acc[0][0] = wmma_f16(af0, bf0, acc[0][0]);
    acc[0][1] = wmma_f16(af0, bf1, acc[0][1]);
    acc[1][0] = wmma_f16(af1, bf0, acc[1][0]);
    acc[1][1] = wmma_f16(af1, bf1, acc[1][1]);
  }

  // epilogue: C fragment row = vgpr + 8*(lane>=16), col = lane&15
#pragma unroll
  for (int i = 0; i < 2; ++i)
#pragma unroll
    for (int j = 0; j < 2; ++j)
#pragma unroll
      for (int e = 0; e < 8; ++e) {
        const int gm = m0 + wr * 32 + i * 16 + lhalf * 8 + e;
        const int gn = n0 + wc * 32 + j * 16 + ln16;
        if (gn < N) {
          float val = acc[i][j][e] + (bias ? bias[gn] : 0.f);
          if constexpr (EPI == 1)
            val = 0.5f * val * (1.f + erff(val * 0.70710678118654752f));
          if constexpr (EPI == 2) val += res[(size_t)gm * N + gn];
          C[(size_t)gm * N + gn] = val;
        }
      }
}

// ---------------------------------------------------------------------------
// Flash attention (causal), head_dim=64. One block per (b, h, 64-query rows).
// 128 threads = 4 waves; wave w owns query rows [w*16, w*16+16).
// Q@K^T and P@V both via v_wmma_f32_16x16x32_f16; online softmax in registers.
// ---------------------------------------------------------------------------
__global__ __launch_bounds__(128) void flash_attn_kernel(
    const float* __restrict__ q, const float* __restrict__ k,
    const float* __restrict__ v, float* __restrict__ o) {
  using namespace cfg;
  const int qblk = blockIdx.x, hh = blockIdx.y, bb = blockIdx.z;
  const int tid = threadIdx.x;
  const int wave = tid >> 5, lane = tid & 31;
  const int ln16 = lane & 15, lhalf = lane >> 4;

  __shared__ _Float16 Qs[64 * 64];
  __shared__ _Float16 Ks[64 * 64];
  __shared__ _Float16 Vs[64 * 64];
  __shared__ _Float16 Ps[4][16 * 64];

  const size_t hbase = (size_t)bb * S * D + (size_t)hh * HD;

  for (int idx = tid; idx < 64 * 64; idx += 128) {
    const int r = idx >> 6, c = idx & 63;
    Qs[idx] = (_Float16)q[hbase + (size_t)(qblk * 64 + r) * D + c];
  }
  __syncthreads();

  const v16h qa0 = load_a_frag(Qs, wave * 16, 0, 64);
  const v16h qa1 = load_a_frag(Qs, wave * 16, 32, 64);

  v8f acc[4];
  float mrow[8], lrow[8];
#pragma unroll
  for (int j = 0; j < 4; ++j)
#pragma unroll
    for (int e = 0; e < 8; ++e) acc[j][e] = 0.f;
#pragma unroll
  for (int e = 0; e < 8; ++e) { mrow[e] = -1e30f; lrow[e] = 0.f; }

  const int grow_base = qblk * 64 + wave * 16 + lhalf * 8;

  for (int kb = 0; kb <= qblk; ++kb) {
    __syncthreads();  // previous iteration done with Ks/Vs
    for (int idx = tid; idx < 64 * 64; idx += 128) {
      const int r = idx >> 6, c = idx & 63;
      const size_t off = hbase + (size_t)(kb * 64 + r) * D + c;
      Ks[idx] = (_Float16)k[off];
      Vs[idx] = (_Float16)v[off];
    }
    __syncthreads();

    // S tile (16 rows x 64 keys) = 4 WMMA accumulators
    v8f st[4];
#pragma unroll
    for (int j = 0; j < 4; ++j) {
      v8f z;
#pragma unroll
      for (int e = 0; e < 8; ++e) z[e] = 0.f;
      z = wmma_f16(qa0, load_bT_frag(Ks, 0, j * 16, 64), z);
      z = wmma_f16(qa1, load_bT_frag(Ks, 32, j * 16, 64), z);
      st[j] = z;
    }

    // scale (HD^-0.5 = 1/8) + causal mask
#pragma unroll
    for (int j = 0; j < 4; ++j) {
      const int gcol = kb * 64 + j * 16 + ln16;
#pragma unroll
      for (int e = 0; e < 8; ++e) {
        float sv = st[j][e] * 0.125f;
        if (gcol > grow_base + e) sv = -1e30f;
        st[j][e] = sv;
      }
    }

    // per-row max: cols live across the 16 lanes of a half-wave
    float nm[8];
#pragma unroll
    for (int e = 0; e < 8; ++e) {
      float m0 = fmaxf(fmaxf(st[0][e], st[1][e]), fmaxf(st[2][e], st[3][e]));
      m0 = fmaxf(m0, __shfl_xor(m0, 1, 32));
      m0 = fmaxf(m0, __shfl_xor(m0, 2, 32));
      m0 = fmaxf(m0, __shfl_xor(m0, 4, 32));
      m0 = fmaxf(m0, __shfl_xor(m0, 8, 32));
      nm[e] = m0;
    }

    // online-softmax rescale
#pragma unroll
    for (int e = 0; e < 8; ++e) {
      const float newm = fmaxf(mrow[e], nm[e]);
      const float corr = __expf(mrow[e] - newm);
      mrow[e] = newm;
      lrow[e] *= corr;
#pragma unroll
      for (int j = 0; j < 4; ++j) acc[j][e] *= corr;
    }

    // P = exp(S - m), accumulate row sums
#pragma unroll
    for (int j = 0; j < 4; ++j)
#pragma unroll
      for (int e = 0; e < 8; ++e) st[j][e] = __expf(st[j][e] - mrow[e]);
#pragma unroll
    for (int e = 0; e < 8; ++e) {
      float rs = st[0][e] + st[1][e] + st[2][e] + st[3][e];
      rs += __shfl_xor(rs, 1, 32);
      rs += __shfl_xor(rs, 2, 32);
      rs += __shfl_xor(rs, 4, 32);
      rs += __shfl_xor(rs, 8, 32);
      lrow[e] += rs;
    }

    // bounce P through LDS to re-lay as an A-fragment (C-layout -> A-layout)
#pragma unroll
    for (int j = 0; j < 4; ++j)
#pragma unroll
      for (int e = 0; e < 8; ++e)
        Ps[wave][(e + lhalf * 8) * 64 + j * 16 + ln16] = (_Float16)st[j][e];
    __syncthreads();

    const v16h pa0 = load_a_frag(Ps[wave], 0, 0, 64);
    const v16h pa1 = load_a_frag(Ps[wave], 0, 32, 64);
#pragma unroll
    for (int j = 0; j < 4; ++j) {
      acc[j] = wmma_f16(pa0, load_b_frag(Vs, 0, j * 16, 64), acc[j]);
      acc[j] = wmma_f16(pa1, load_b_frag(Vs, 32, j * 16, 64), acc[j]);
    }
  }

  // o[b, row, h*64 + col] = acc / l
#pragma unroll
  for (int j = 0; j < 4; ++j)
#pragma unroll
    for (int e = 0; e < 8; ++e) {
      const int row = grow_base + e;
      o[hbase + (size_t)row * D + j * 16 + ln16] = acc[j][e] / lrow[e];
    }
}

// ---------------------------------------------------------------------------
// Host orchestration
// ---------------------------------------------------------------------------
extern "C" void kernel_launch(void* const* d_in, const int* in_sizes, int n_in,
                              void* d_out, int out_size, void* d_ws,
                              size_t ws_size, hipStream_t stream) {
  using namespace cfg;
  (void)in_sizes; (void)n_in; (void)out_size; (void)ws_size;

  const int*   ids  = (const int*)  d_in[0];
  const float* tok  = (const float*)d_in[1];
  const float* pos  = (const float*)d_in[2];
  const float* Wq   = (const float*)d_in[3];
  const float* bq   = (const float*)d_in[4];
  const float* Wk   = (const float*)d_in[5];
  const float* bk   = (const float*)d_in[6];
  const float* Wv   = (const float*)d_in[7];
  const float* bv   = (const float*)d_in[8];
  const float* Wo   = (const float*)d_in[9];
  const float* bo   = (const float*)d_in[10];
  const float* ln1s = (const float*)d_in[11];
  const float* ln1b = (const float*)d_in[12];
  const float* ln2s = (const float*)d_in[13];
  const float* ln2b = (const float*)d_in[14];
  const float* W1   = (const float*)d_in[15];
  const float* b1   = (const float*)d_in[16];
  const float* W2   = (const float*)d_in[17];
  const float* b2   = (const float*)d_in[18];
  const float* lnfs = (const float*)d_in[19];
  const float* lnfb = (const float*)d_in[20];
  float* out = (float*)d_out;

  // workspace layout (floats): h | n | q | k | v | o ; f1 reuses q..o (4MB*4)
  const size_t RD = (size_t)MROWS * D;  // 1M floats
  float* ws = (float*)d_ws;
  float* h   = ws;
  float* nrm = h + RD;
  float* qb  = nrm + RD;
  float* kb  = qb + RD;
  float* vb  = kb + RD;
  float* ob  = vb + RD;
  float* f1  = qb;  // 4*RD floats, reused after attention consumes q/k/v/o

  embed_kernel<<<MROWS, D, 0, stream>>>(ids, tok, pos, h);

  const dim3 gD(D / 64, MROWS / 64);        // 4 x 64
  const dim3 gF(4 * D / 64, MROWS / 64);    // 16 x 64
  const dim3 gA(S / 64, H, B);              // 32 x 4 x 2

  for (int l = 0; l < L; ++l) {
    const size_t wDD = (size_t)l * D * D;
    const size_t wDF = (size_t)l * D * 4 * D;
    layernorm_kernel<<<MROWS, D, 0, stream>>>(h, ln1s + l * D, ln1b + l * D, nrm);
    gemm_kernel<false, 0><<<gD, 128, 0, stream>>>(nrm, Wq + wDD, bq + l * D,
                                                  nullptr, qb, MROWS, D, D);
    gemm_kernel<false, 0><<<gD, 128, 0, stream>>>(nrm, Wk + wDD, bk + l * D,
                                                  nullptr, kb, MROWS, D, D);
    gemm_kernel<false, 0><<<gD, 128, 0, stream>>>(nrm, Wv + wDD, bv + l * D,
                                                  nullptr, vb, MROWS, D, D);
    flash_attn_kernel<<<gA, 128, 0, stream>>>(qb, kb, vb, ob);
    gemm_kernel<false, 2><<<gD, 128, 0, stream>>>(ob, Wo + wDD, bo + l * D, h,
                                                  h, MROWS, D, D);
    layernorm_kernel<<<MROWS, D, 0, stream>>>(h, ln2s + l * D, ln2b + l * D, nrm);
    gemm_kernel<false, 1><<<gF, 128, 0, stream>>>(nrm, W1 + wDF, b1 + l * 4 * D,
                                                  nullptr, f1, MROWS, 4 * D, D);
    gemm_kernel<false, 2><<<gD, 128, 0, stream>>>(f1, W2 + wDF, b2 + l * D, h,
                                                  h, MROWS, D, 4 * D);
  }

  layernorm_kernel<<<MROWS, D, 0, stream>>>(h, lnfs, lnfb, nrm);
  const dim3 gV((V + 63) / 64, MROWS / 64); // 786 x 64
  gemm_kernel<true, 0><<<gV, 128, 0, stream>>>(nrm, tok, nullptr, nullptr, out,
                                               MROWS, V, D);
}
